// CausalSelfAttention_12816182411588
// MI455X (gfx1250) — matrix operand order repaired and checked by reading, not compiled, not run on hardware
//
#include <hip/hip_runtime.h>
#include <hip/hip_bf16.h>

// ---------------------------------------------------------------------------
// CausalSelfAttention for MI455X (gfx1250), wave32 + WMMA bf16 + async-to-LDS.
//   B=4, T=2048, C=1024, H=16, D=64
// Pipeline:
//   cvt (fp32->bf16 once) -> qkv_gemm (WMMA, dbl-buffered async LDS)
//   -> flash attention (WMMA, dbl-buffered async LDS)
//   -> proj_gemm (WMMA, dbl-buffered async LDS)
// Workspace (bf16): x 16MB, Wqkv 6MB, Wproj 2MB, Q/K/V 16MB ea, O 16MB = 88MB
// ---------------------------------------------------------------------------

typedef __attribute__((ext_vector_type(16))) __bf16 v16bf;
typedef __attribute__((ext_vector_type(8)))  __bf16 bf16x8;
typedef __attribute__((ext_vector_type(8)))  float  v8f;
typedef __attribute__((ext_vector_type(4)))  int    v4i;

#define WMMA_BF16(a, b, c) \
  __builtin_amdgcn_wmma_f32_16x16x32_bf16(false, (a), false, (b), (short)0, (c), false, false)

// ---- CDNA5 async-to-LDS path (ASYNCcnt). Builtin signature (probe-confirmed
// by round-2 diagnostic): (v4i AS1*, v4i AS3*, imm offset, imm cpol).
// Address-space casts go through integers: flat global == AS1 numerically;
// flat LDS low 32 bits == LDS offset (ISA aperture mapping), which is what a
// 32-bit AS3 pointer holds.
typedef __attribute__((address_space(1))) v4i v4i_g;
typedef __attribute__((address_space(3))) v4i v4i_l;

#if defined(__has_builtin) && __has_builtin(__builtin_amdgcn_global_load_async_to_lds_b128)
static __device__ __forceinline__ void async_cp_b128(const __bf16* g, __bf16* l) {
  __builtin_amdgcn_global_load_async_to_lds_b128(
      (v4i_g*)(unsigned long long)g,
      (v4i_l*)(unsigned)(unsigned long long)l, 0, 0);
}
#else
static __device__ __forceinline__ void async_cp_b128(const __bf16* g, __bf16* l) {
  unsigned long long ga = (unsigned long long)g;
  unsigned lofs = (unsigned)(unsigned long long)l;  // flat addr low 32b = LDS offset
  asm volatile("global_load_async_to_lds_b128 %0, %1, off"
               :: "v"(lofs), "v"(ga) : "memory");
}
#endif

#if defined(__has_builtin) && __has_builtin(__builtin_amdgcn_s_wait_asynccnt)
#define WAIT_ASYNC(n) __builtin_amdgcn_s_wait_asynccnt(n)
#else
#define WAIT_ASYNC(n) asm volatile("s_wait_asynccnt " #n ::: "memory")
#endif

static __device__ __forceinline__ v16bf join16(bf16x8 lo, bf16x8 hi) {
  v16bf r;
#pragma unroll
  for (int i = 0; i < 8; ++i) { r[i] = lo[i]; r[i + 8] = hi[i]; }
  return r;
}

// ---------------------------------------------------------------------------
// Kernel 0: fp32 -> bf16 streaming convert (8 elems/thread).
// ---------------------------------------------------------------------------
__global__ __launch_bounds__(256) void cvt_bf16(
    const float* __restrict__ src, __bf16* __restrict__ dst, int n) {
  const int i = (blockIdx.x * 256 + threadIdx.x) * 8;
  if (i >= n) return;
  bf16x8 o;
#pragma unroll
  for (int j = 0; j < 8; ++j) o[j] = (__bf16)src[i + j];
  *(bf16x8*)(dst + i) = o;
}

// ---------------------------------------------------------------------------
// GEMM core (shared): C[128x128 WG tile] = A[M,1024] @ B[N,1024]^T (bf16).
// 8 waves, each 32(M) x 64(N). K-step 64, double-buffered async LDS staging.
// ---------------------------------------------------------------------------
static __device__ __forceinline__ void issue_gemm_tile(
    const __bf16* __restrict__ A, const __bf16* __restrict__ Bm,
    __bf16* Xb, __bf16* Wb, int mBase, int nBase, int kb, int tid) {
#pragma unroll
  for (int j = 0; j < 4; ++j) {             // 1024 chunks of 16B per matrix
    const int cid = tid + 256 * j;          // 0..1023
    const int row = cid >> 3;               // 0..127
    const int c8  = (cid & 7) * 8;          // elem offset within 64
    async_cp_b128(A  + (size_t)(mBase + row) * 1024 + kb + c8, Xb + row * 64 + c8);
    async_cp_b128(Bm + (size_t)(nBase + row) * 1024 + kb + c8, Wb + row * 64 + c8);
  }
}

static __device__ __forceinline__ void gemm_compute_tile(
    const __bf16* Xb, const __bf16* Wb, int wm, int wn, int half, int l16,
    v8f acc[2][4]) {
#pragma unroll
  for (int kc = 0; kc < 2; ++kc) {
    v16bf afr[2];
#pragma unroll
    for (int mt = 0; mt < 2; ++mt) {
      const int row = wm + mt * 16 + l16;
      bf16x8 lo = *(const bf16x8*)&Xb[row * 64 + kc * 32 + half * 8];
      bf16x8 hi = *(const bf16x8*)&Xb[row * 64 + kc * 32 + 16 + half * 8];
      afr[mt] = join16(lo, hi);
    }
#pragma unroll
    for (int nt = 0; nt < 4; ++nt) {
      const int col = wn + nt * 16 + l16;
      bf16x8 lo = *(const bf16x8*)&Wb[col * 64 + kc * 32 + half * 8];
      bf16x8 hi = *(const bf16x8*)&Wb[col * 64 + kc * 32 + 16 + half * 8];
      const v16bf bfr = join16(lo, hi);
#pragma unroll
      for (int mt = 0; mt < 2; ++mt) acc[mt][nt] = WMMA_BF16(afr[mt], bfr, acc[mt][nt]);
    }
  }
}

// ---------------------------------------------------------------------------
// Kernel 1: qkv = x @ W_qkv^T + b_qkv -> Q/K/V [B,H,T,D] bf16, Q pre-scaled.
// Grid (24, 64), block 256.
// ---------------------------------------------------------------------------
__global__ __launch_bounds__(256) void qkv_gemm(
    const __bf16* __restrict__ X, const __bf16* __restrict__ W,
    const float* __restrict__ bias,
    __bf16* __restrict__ Qb, __bf16* __restrict__ Kb, __bf16* __restrict__ Vb) {
  __shared__ alignas(16) __bf16 Xs[2 * 128 * 64];
  __shared__ alignas(16) __bf16 Ws[2 * 128 * 64];

  const int tid  = threadIdx.x;
  const int lane = tid & 31, wave = tid >> 5;
  const int half = lane >> 4, l16 = lane & 15;
  const int mBase = blockIdx.y * 128, nBase = blockIdx.x * 128;
  const int wm = (wave & 3) * 32, wn = (wave >> 2) * 64;

  v8f acc[2][4];
#pragma unroll
  for (int mt = 0; mt < 2; ++mt)
#pragma unroll
    for (int nt = 0; nt < 4; ++nt) acc[mt][nt] = v8f{0, 0, 0, 0, 0, 0, 0, 0};

  issue_gemm_tile(X, W, Xs, Ws, mBase, nBase, 0, tid);
  int cur = 0;
  for (int i = 0; i < 16; ++i) {            // 1024 / 64
    if (i + 1 < 16) {
      issue_gemm_tile(X, W, Xs + (cur ^ 1) * 8192, Ws + (cur ^ 1) * 8192,
                      mBase, nBase, (i + 1) * 64, tid);
      WAIT_ASYNC(8);                        // current tile's 8 ops complete
    } else {
      WAIT_ASYNC(0);
    }
    __syncthreads();                        // tile i fully in LDS for all waves
    gemm_compute_tile(Xs + cur * 8192, Ws + cur * 8192, wm, wn, half, l16, acc);
    __syncthreads();                        // all waves done before buffer reuse
    cur ^= 1;
  }

#pragma unroll
  for (int mt = 0; mt < 2; ++mt) {
#pragma unroll
    for (int nt = 0; nt < 4; ++nt) {
      const int n     = nBase + wn + nt * 16 + l16;
      const int which = n >> 10;            // 0=Q 1=K 2=V
      const int c     = n & 1023;
      const int h     = c >> 6, d = c & 63;
      __bf16* dst = (which == 0) ? Qb : (which == 1) ? Kb : Vb;
      const float bv    = bias[n];
      const float scale = (which == 0) ? 0.125f : 1.0f;  // fold 1/sqrt(D) into Q
#pragma unroll
      for (int r = 0; r < 8; ++r) {
        const int m = mBase + wm + mt * 16 + r + half * 8;
        const int b = m >> 11, t = m & 2047;
        const float v = (acc[mt][nt][r] + bv) * scale;
        dst[(size_t)((b * 16 + h) * 2048 + t) * 64 + d] = (__bf16)v;
      }
    }
  }
}

// ---------------------------------------------------------------------------
// Kernel 2: flash attention. Grid (T/64, B*H) = (32, 64), block 128 (4 waves).
// Double-buffered async K/V staging; V transposed LDS->LDS for B-operand.
// ---------------------------------------------------------------------------
static __device__ __forceinline__ void issue_attn_tile(
    const __bf16* __restrict__ Kh, const __bf16* __restrict__ Vh,
    __bf16* Kbuf, __bf16* Vbuf, int kb, int tid) {
#pragma unroll
  for (int j = 0; j < 2; ++j) {             // 256 chunks of 16B per matrix
    const int cid = tid + 128 * j;          // 0..255
    const int row = cid >> 3;               // 0..31
    const int c8  = (cid & 7) * 8;
    async_cp_b128(Kh + (size_t)(kb + row) * 64 + c8, Kbuf + row * 64 + c8);
    async_cp_b128(Vh + (size_t)(kb + row) * 64 + c8, Vbuf + row * 64 + c8);
  }
}

__global__ __launch_bounds__(128) void attn(
    const __bf16* __restrict__ Qb, const __bf16* __restrict__ Kb,
    const __bf16* __restrict__ Vb, __bf16* __restrict__ Ob) {
  __shared__ alignas(16) __bf16 Ks[2 * 32 * 64];  // [buf][key][d]
  __shared__ alignas(16) __bf16 Vr[2 * 32 * 64];  // [buf][key][d] row-major
  __shared__ alignas(16) __bf16 Vt[64 * 32];      // [d][key] transposed
  __shared__ alignas(16) __bf16 Ps[4 * 16 * 32];  // per-wave P tile [q][key]

  const int tid  = threadIdx.x;
  const int lane = tid & 31, wave = tid >> 5;
  const int half = lane >> 4, l16 = lane & 15;
  const int bh = blockIdx.y;
  const int b  = bh >> 4, h = bh & 15;
  const int qbase = blockIdx.x * 64;
  const size_t headOff = (size_t)bh * 2048 * 64;
  const __bf16* Qh = Qb + headOff;
  const __bf16* Kh = Kb + headOff;
  const __bf16* Vh = Vb + headOff;

  // Q fragments (A-layout), contraction D=64 as two K=32 chunks. Scale folded.
  v16bf qf[2];
  {
    const int row = qbase + wave * 16 + l16;
    const __bf16* qp = Qh + (size_t)row * 64;
#pragma unroll
    for (int kc = 0; kc < 2; ++kc) {
      bf16x8 lo = *(const bf16x8*)(qp + kc * 32 + half * 8);
      bf16x8 hi = *(const bf16x8*)(qp + kc * 32 + 16 + half * 8);
      qf[kc] = join16(lo, hi);
    }
  }

  v8f o[4];
#pragma unroll
  for (int nt = 0; nt < 4; ++nt) o[nt] = v8f{0, 0, 0, 0, 0, 0, 0, 0};
  float mrun[8], lrun[8];
#pragma unroll
  for (int r = 0; r < 8; ++r) { mrun[r] = -3.0e38f; lrun[r] = 0.0f; }

  __bf16* Pw = &Ps[wave * 16 * 32];
  const int kend = qbase + 64;

  issue_attn_tile(Kh, Vh, Ks, Vr, 0, tid);
  int cur = 0;
  for (int kb = 0; kb < kend; kb += 32) {
    if (kb + 32 < kend) {
      issue_attn_tile(Kh, Vh, Ks + (cur ^ 1) * 2048, Vr + (cur ^ 1) * 2048,
                      kb + 32, tid);
      WAIT_ASYNC(4);
    } else {
      WAIT_ASYNC(0);
    }
    __syncthreads();  // tile kb complete in Ks[cur], Vr[cur]; prev O-updates done

    // Transpose V block into Vt (readers are after the next barrier).
    {
      const __bf16* Vc = Vr + cur * 2048;
#pragma unroll
      for (int j = 0; j < 2; ++j) {
        const int cid = tid + 128 * j;
        const int row = cid >> 3;
        const int c8  = (cid & 7) * 8;
        bf16x8 v = *(const bf16x8*)&Vc[row * 64 + c8];
#pragma unroll
        for (int ii = 0; ii < 8; ++ii) Vt[(c8 + ii) * 32 + row] = v[ii];
      }
    }

    // S = Q K^T for two 16-key tiles.
    const __bf16* Kc = Ks + cur * 2048;
    v8f s0 = v8f{0, 0, 0, 0, 0, 0, 0, 0};
    v8f s1 = v8f{0, 0, 0, 0, 0, 0, 0, 0};
#pragma unroll
    for (int kc = 0; kc < 2; ++kc) {
      {
        const int key = l16;
        bf16x8 lo = *(const bf16x8*)&Kc[key * 64 + kc * 32 + half * 8];
        bf16x8 hi = *(const bf16x8*)&Kc[key * 64 + kc * 32 + 16 + half * 8];
        s0 = WMMA_BF16(qf[kc], join16(lo, hi), s0);
      }
      {
        const int key = 16 + l16;
        bf16x8 lo = *(const bf16x8*)&Kc[key * 64 + kc * 32 + half * 8];
        bf16x8 hi = *(const bf16x8*)&Kc[key * 64 + kc * 32 + 16 + half * 8];
        s1 = WMMA_BF16(qf[kc], join16(lo, hi), s1);
      }
    }

    // Causal mask + online softmax (row lives across 16 lanes of a half).
    float alpha[8];
#pragma unroll
    for (int r = 0; r < 8; ++r) {
      const int qrow = qbase + wave * 16 + r + half * 8;
      const int k0 = kb + l16, k1 = kb + 16 + l16;
      float v0 = (k0 <= qrow) ? s0[r] : -3.0e38f;
      float v1 = (k1 <= qrow) ? s1[r] : -3.0e38f;
      float rm = fmaxf(v0, v1);
      rm = fmaxf(rm, __shfl_xor(rm, 1));
      rm = fmaxf(rm, __shfl_xor(rm, 2));
      rm = fmaxf(rm, __shfl_xor(rm, 4));
      rm = fmaxf(rm, __shfl_xor(rm, 8));
      const float mn = fmaxf(mrun[r], rm);
      const float al = __expf(mrun[r] - mn);
      const float p0 = __expf(v0 - mn);
      const float p1 = __expf(v1 - mn);
      float rs = p0 + p1;
      rs += __shfl_xor(rs, 1);
      rs += __shfl_xor(rs, 2);
      rs += __shfl_xor(rs, 4);
      rs += __shfl_xor(rs, 8);
      lrun[r] = lrun[r] * al + rs;
      mrun[r] = mn;
      alpha[r] = al;
      const int prow = r + half * 8;          // P (C-layout) -> LDS [q][key]
      Pw[prow * 32 + l16]      = (__bf16)p0;
      Pw[prow * 32 + 16 + l16] = (__bf16)p1;
    }
    __syncthreads();  // P + Vt visible; Ks/Vr[cur] no longer needed afterwards

    // P fragment (A-layout); O = O*alpha + P V.
    v16bf pf;
    {
      bf16x8 lo = *(const bf16x8*)&Pw[l16 * 32 + half * 8];
      bf16x8 hi = *(const bf16x8*)&Pw[l16 * 32 + 16 + half * 8];
      pf = join16(lo, hi);
    }
#pragma unroll
    for (int nt = 0; nt < 4; ++nt) {
#pragma unroll
      for (int r = 0; r < 8; ++r) o[nt][r] *= alpha[r];
      const int dcol = nt * 16 + l16;
      bf16x8 lo = *(const bf16x8*)&Vt[dcol * 32 + half * 8];
      bf16x8 hi = *(const bf16x8*)&Vt[dcol * 32 + 16 + half * 8];
      o[nt] = WMMA_BF16(pf, join16(lo, hi), o[nt]);
    }
    cur ^= 1;
  }

  // Normalize, store to O [B*T, C] bf16 (c = h*64 + d).
#pragma unroll
  for (int nt = 0; nt < 4; ++nt) {
#pragma unroll
    for (int r = 0; r < 8; ++r) {
      const int trow = qbase + wave * 16 + r + half * 8;
      const float v = o[nt][r] / lrun[r];
      Ob[(size_t)(b * 2048 + trow) * 1024 + h * 64 + nt * 16 + l16] = (__bf16)v;
    }
  }
}

// ---------------------------------------------------------------------------
// Kernel 3: out = O @ W_proj^T + b_proj (fp32 out). Grid (8, 64), block 256.
// ---------------------------------------------------------------------------
__global__ __launch_bounds__(256) void proj_gemm(
    const __bf16* __restrict__ A, const __bf16* __restrict__ W,
    const float* __restrict__ bias, float* __restrict__ out) {
  __shared__ alignas(16) __bf16 Xs[2 * 128 * 64];
  __shared__ alignas(16) __bf16 Ws[2 * 128 * 64];

  const int tid  = threadIdx.x;
  const int lane = tid & 31, wave = tid >> 5;
  const int half = lane >> 4, l16 = lane & 15;
  const int mBase = blockIdx.y * 128, nBase = blockIdx.x * 128;
  const int wm = (wave & 3) * 32, wn = (wave >> 2) * 64;

  v8f acc[2][4];
#pragma unroll
  for (int mt = 0; mt < 2; ++mt)
#pragma unroll
    for (int nt = 0; nt < 4; ++nt) acc[mt][nt] = v8f{0, 0, 0, 0, 0, 0, 0, 0};

  issue_gemm_tile(A, W, Xs, Ws, mBase, nBase, 0, tid);
  int cur = 0;
  for (int i = 0; i < 16; ++i) {
    if (i + 1 < 16) {
      issue_gemm_tile(A, W, Xs + (cur ^ 1) * 8192, Ws + (cur ^ 1) * 8192,
                      mBase, nBase, (i + 1) * 64, tid);
      WAIT_ASYNC(8);
    } else {
      WAIT_ASYNC(0);
    }
    __syncthreads();
    gemm_compute_tile(Xs + cur * 8192, Ws + cur * 8192, wm, wn, half, l16, acc);
    __syncthreads();
    cur ^= 1;
  }

#pragma unroll
  for (int mt = 0; mt < 2; ++mt) {
#pragma unroll
    for (int nt = 0; nt < 4; ++nt) {
      const int n  = nBase + wn + nt * 16 + l16;
      const float bv = bias[n];
#pragma unroll
      for (int r = 0; r < 8; ++r) {
        const int m = mBase + wm + mt * 16 + r + half * 8;
        out[(size_t)m * 1024 + n] = acc[mt][nt][r] + bv;
      }
    }
  }
}

// ---------------------------------------------------------------------------
extern "C" void kernel_launch(void* const* d_in, const int* in_sizes, int n_in,
                              void* d_out, int out_size, void* d_ws, size_t ws_size,
                              hipStream_t stream) {
  const float* x     = (const float*)d_in[0];
  const float* Wqkv  = (const float*)d_in[1];
  const float* bqkv  = (const float*)d_in[2];
  const float* Wproj = (const float*)d_in[3];
  const float* bproj = (const float*)d_in[4];
  float* out = (float*)d_out;

  const size_t N_X    = (size_t)8192 * 1024;   // 8,388,608
  const size_t N_WQKV = (size_t)3072 * 1024;   // 3,145,728
  const size_t N_WPRJ = (size_t)1024 * 1024;   // 1,048,576

  __bf16* Xb     = (__bf16*)d_ws;
  __bf16* Wqkvb  = Xb + N_X;
  __bf16* Wprojb = Wqkvb + N_WQKV;
  __bf16* Qb     = Wprojb + N_WPRJ;
  __bf16* Kb     = Qb + N_X;
  __bf16* Vb     = Kb + N_X;
  __bf16* Ob     = Vb + N_X;

  // 0) one-time fp32 -> bf16 converts (streaming, ~2.5us at HBM peak)
  cvt_bf16<<<dim3(N_X / 2048), 256, 0, stream>>>(x, Xb, (int)N_X);
  cvt_bf16<<<dim3(N_WQKV / 2048), 256, 0, stream>>>(Wqkv, Wqkvb, (int)N_WQKV);
  cvt_bf16<<<dim3(N_WPRJ / 2048), 256, 0, stream>>>(Wproj, Wprojb, (int)N_WPRJ);
  // 1) QKV projection -> Q/K/V [B,H,T,D] bf16
  qkv_gemm<<<dim3(24, 64), 256, 0, stream>>>(Xb, Wqkvb, bqkv, Qb, Kb, Vb);
  // 2) Flash attention -> O [B*T, C] bf16
  attn<<<dim3(32, 64), 128, 0, stream>>>(Qb, Kb, Vb, Ob);
  // 3) Output projection -> fp32 out
  proj_gemm<<<dim3(8, 64), 256, 0, stream>>>(Ob, Wprojb, bproj, out);
}
